// SparseAttnIndexer_90134183673904
// MI455X (gfx1250) — compile-verified, matrix operand order checked by reading, not run on hardware
//
#include <hip/hip_runtime.h>
#include <hip/hip_bf16.h>

// ---------------------------------------------------------------------------
// SparseAttnIndexer decode: paged KV gather + bf16 WMMA logits + head-weighted
// reduce + per-row top-2048 select.  Shapes fixed by the harness:
//   B=64, H=64, D=128, BLOCK=64, MAX_BLOCKS=128, S=8192, TOPK=2048
// Memory-bound on the 128 MiB KV stream -> K read exactly once into registers
// (saddr-form b128 clauses), Q staged into LDS via async direct-to-LDS loads,
// v_wmma_f32_16x16x32_bf16 for the logits.
// ---------------------------------------------------------------------------

typedef __attribute__((ext_vector_type(16))) __bf16 v16bf;
typedef __attribute__((ext_vector_type(8)))  float  v8f;

#define B_SZ        64
#define H_SZ        64
#define D_SZ        128
#define S_SZ        8192
#define MAXBLK      128
#define TOPK        2048
#define CHUNK_KEYS  1024          // keys per workgroup
#define NCHUNK      (S_SZ / CHUNK_KEYS)
#define NEG_MAX_F32 (-3.4028234663852886e38f)

// ---------------------------------------------------------------------------
// Kernel 1: weighted[b, s] = sum_h w[b,h] * (q[b,h,:] . k[b,s,:])  (fp32 acc)
// grid = (NCHUNK, B), block = 256 (8 waves of 32)
// ---------------------------------------------------------------------------
__global__ __launch_bounds__(256) void
indexer_weighted_kernel(const __bf16* __restrict__ q,        // [B,H,D]
                        const __bf16* __restrict__ wgt,      // [B,H]
                        const __bf16* __restrict__ kv,       // [NUM_BLOCKS,64,D]
                        const int*    __restrict__ btab,     // [B,MAXBLK]
                        const int*    __restrict__ slen,     // [B]
                        float*        __restrict__ wout)     // [B,S]
{
    __shared__ __align__(32) __bf16 qs[H_SZ * D_SZ];   // 16 KiB: Q for this b
    __shared__ float wf[H_SZ];

    const int b     = blockIdx.y;
    const int chunk = blockIdx.x;
    const int tid   = threadIdx.x;
    const int lane  = tid & 31;
    // force wave id scalar so all tile addressing becomes SGPR math
    const int wv    = __builtin_amdgcn_readfirstlane(tid >> 5);

    // --- stage Q[b] into LDS with async direct-to-LDS b128 loads ------------
    {
        const __bf16*  qsrc = q + (size_t)b * H_SZ * D_SZ;
        const unsigned lbase = (unsigned)(uintptr_t)qs;      // LDS byte offset
#pragma unroll
        for (int k = 0; k < 4; ++k) {                        // 4 x 16B / thread
            const int i = tid + k * 256;
            const unsigned long long gaddr =
                (unsigned long long)(uintptr_t)qsrc + (unsigned)(i * 16);
            const unsigned laddr = lbase + (unsigned)(i * 16);
            asm volatile("global_load_async_to_lds_b128 %0, %1, off"
                         :: "v"(laddr), "v"(gaddr) : "memory");
        }
        if (tid < H_SZ) wf[tid] = (float)wgt[(size_t)b * H_SZ + tid];
        asm volatile("s_wait_asynccnt 0x0" ::: "memory");    // our async done
    }
    __syncthreads();                                         // cross-wave

    const int hi = lane >> 4;           // which 16-dim half this lane carries
    const int n  = lane & 15;           // key column within the 16-key tile
    const int lane_byte = n * (D_SZ * 2) + hi * 32;          // per-lane offset

    // per-lane cache of the 32 head-weights this lane reduces against
    float wc[4][8];
#pragma unroll
    for (int ht = 0; ht < 4; ++ht)
#pragma unroll
        for (int r = 0; r < 8; ++r)
            wc[ht][r] = wf[ht * 16 + r + 8 * hi];

    const int  sl    = slen[b];
    const int* btrow = btab + (size_t)b * MAXBLK;
    const int  s_wg  = chunk * CHUNK_KEYS;

    // each wave owns 8 tiles of 16 keys
    for (int t = 0; t < 8; ++t) {
        const int s0  = s_wg + wv * 128 + t * 16;            // uniform
        const int blk = __builtin_amdgcn_readfirstlane(btrow[s0 >> 6]);
        const __bf16* kbase = kv + ((size_t)blk * 64 + (s0 & 63)) * D_SZ;
        const __bf16* krow  = (const __bf16*)((const char*)kbase + lane_byte);

        // prefetch next tile's stream (-> global_prefetch_b8)
        __builtin_prefetch(krow + 16 * D_SZ, 0, 1);

        // B-operand fragments: 4 x (32-dim slice), 16 bf16 per lane each
        v16bf bfrag[4];
#pragma unroll
        for (int dt = 0; dt < 4; ++dt)
            bfrag[dt] = *(const v16bf*)(krow + dt * 32);

        float partial = 0.0f;
#pragma unroll
        for (int ht = 0; ht < 4; ++ht) {
            v8f acc = {};
#pragma unroll
            for (int dt = 0; dt < 4; ++dt) {
                const v16bf afrag = *(const v16bf*)
                    (qs + (size_t)(ht * 16 + n) * D_SZ + dt * 32 + hi * 16);
                acc = __builtin_amdgcn_wmma_f32_16x16x32_bf16(
                        /*neg_a=*/false, afrag,
                        /*neg_b=*/false, bfrag[dt],
                        /*c_mod=*/(short)0, acc,
                        /*reuse_a=*/false, /*reuse_b=*/false);
            }
            // lane holds rows M = r + 8*hi of the 16x16 logit tile, col n
#pragma unroll
            for (int r = 0; r < 8; ++r) partial += wc[ht][r] * acc[r];
        }
        // fold the two row-halves (lane l <-> l^16 share key column n)
        partial += __shfl_xor(partial, 16, 32);

        if (lane < 16) {
            const int s = s0 + lane;
            wout[(size_t)b * S_SZ + s] = (s < sl) ? partial : NEG_MAX_F32;
        }
    }
}

// ---------------------------------------------------------------------------
// Kernel 2: per-row top-2048 by descending value (ties -> lower index first).
// Full bitonic sort of 8192 packed 64-bit keys in LDS. grid = B, block = 512.
// key = (sortable_u32(value) << 32) | ~index
// ---------------------------------------------------------------------------
__global__ __launch_bounds__(512) void
indexer_topk_kernel(const float* __restrict__ wout, int* __restrict__ iout)
{
    __shared__ unsigned long long skey[S_SZ];   // 64 KiB

    const int b   = blockIdx.x;
    const int tid = threadIdx.x;

    for (int i = tid; i < S_SZ; i += 512) {
        unsigned u = __float_as_uint(wout[(size_t)b * S_SZ + i]);
        u ^= (unsigned)((int)u >> 31) | 0x80000000u;     // order-preserving map
        skey[i] = ((unsigned long long)u << 32) | (unsigned)(~i);
    }

    for (unsigned k = 2; k <= S_SZ; k <<= 1) {
        for (unsigned j = k >> 1; j > 0; j >>= 1) {
            __syncthreads();
            for (unsigned i = tid; i < S_SZ; i += 512) {
                const unsigned ixj = i ^ j;
                if (ixj > i) {
                    const unsigned long long a = skey[i];
                    const unsigned long long c = skey[ixj];
                    const bool desc = ((i & k) == 0);    // k==S_SZ -> all desc
                    if (desc ? (a < c) : (a > c)) {
                        skey[i]   = c;
                        skey[ixj] = a;
                    }
                }
            }
        }
    }
    __syncthreads();

    for (int i = tid; i < TOPK; i += 512)
        iout[(size_t)b * TOPK + i] = (int)(~(unsigned)skey[i]);
}

// ---------------------------------------------------------------------------
extern "C" void kernel_launch(void* const* d_in, const int* in_sizes, int n_in,
                              void* d_out, int out_size, void* d_ws, size_t ws_size,
                              hipStream_t stream)
{
    (void)in_sizes; (void)n_in; (void)d_ws; (void)ws_size; (void)out_size;

    const __bf16* q    = (const __bf16*)d_in[0];
    const __bf16* wgt  = (const __bf16*)d_in[1];
    const __bf16* kv   = (const __bf16*)d_in[2];
    const int*    btab = (const int*)d_in[3];
    const int*    slen = (const int*)d_in[4];
    // d_in[5] = topk_tokens scalar (fixed at 2048 by the harness shapes)

    int*   iout = (int*)d_out;                          // [B, TOPK] int32
    float* wout = (float*)d_out + (size_t)B_SZ * TOPK;  // [B, S] fp32

    indexer_weighted_kernel<<<dim3(NCHUNK, B_SZ), 256, 0, stream>>>(
        q, wgt, kv, btab, slen, wout);

    indexer_topk_kernel<<<dim3(B_SZ), 512, 0, stream>>>(wout, iout);
}
